// E_GCL_40896678592759
// MI455X (gfx1250) — matrix operand order repaired and checked by reading, not compiled, not run on hardware
//
#include <hip/hip_runtime.h>
#include <hip/hip_bf16.h>

// ---------------------------------------------------------------------------
// E_GCL layer for MI455X (gfx1250).  B=8, N=256, D=H=128.
// Edge MLP decomposed so the heavy work is two [E,128]x[128,128] GEMMs
// executed with v_wmma_f32_16x16x32_f16 (f16 in, f32 accumulate).
// Round 3: fast-rcp SiLU (kills v_div_scale chains), biases via LDS instead
// of dynamically-indexed register arrays (kills v_movrels/m0 in inner loop),
// agg accumulation via per-ct row-sum + ds_add_f32.
// ---------------------------------------------------------------------------

typedef __attribute__((ext_vector_type(16))) _Float16 v16h;
typedef __attribute__((ext_vector_type(8)))  _Float16 v8h;
typedef __attribute__((ext_vector_type(4)))  _Float16 v4h;
typedef __attribute__((ext_vector_type(8)))  float    v8f;

#define BATCH 8
#define NN    256
#define DD    128
#define PADF  136   // padded f16 row stride for LDS tiles (bank-conflict relief)

__device__ __forceinline__ float silu_f(float v) {
    // x * rcp(1+exp(-x)) : v_exp_f32 + v_rcp_f32, no IEEE-div fixup chain
    return v * __builtin_amdgcn_rcpf(1.0f + __expf(-v));
}

__device__ __forceinline__ v16h combine16(v8h lo, v8h hi) {
    v16h r;
#pragma unroll
    for (int i = 0; i < 8; ++i) { r[i] = lo[i]; r[i + 8] = hi[i]; }
    return r;
}

// ---------------------------------------------------------------------------
// Kernel 1: per-node projections  Apre = h @ we1[0:128], Bpre = h @ we1[128:256]
// ---------------------------------------------------------------------------
__global__ __launch_bounds__(128) void egcl_pre_kernel(
    const float* __restrict__ h, const float* __restrict__ we1,
    float* __restrict__ Apre, float* __restrict__ Bpre)
{
    __shared__ float s_h[DD];
    const int row = blockIdx.x;       // b*N + n
    const int f   = threadIdx.x;
    s_h[f] = h[row * DD + f];
    __syncthreads();
    float a = 0.f, b = 0.f;
#pragma unroll 8
    for (int k = 0; k < DD; ++k) {
        const float hv = s_h[k];
        a += hv * we1[k * DD + f];
        b += hv * we1[(DD + k) * DD + f];
    }
    Apre[row * DD + f] = a;
    Bpre[row * DD + f] = b;
}

// ---------------------------------------------------------------------------
// Kernel 2: repack we2 / wc1 (row-major [K=128, N=128] f32) into WMMA
// B-fragment order (f16).  Fragment (ct,kf) covers cols ct*16..+15,
// K = kf*32..+31.  Per the CDNA5 16-bit B (32x16) layout:
//   lanes 0-15  : column = lane,      halves h -> K = 32*kf + (h<8 ? h : 16+h-8)
//   lanes 16-31 : column = lane-16,   halves h -> K = 32*kf + 8 + (h<8 ? h : 16+h-8)
// Stored lane-contiguous: dst[((ct*4+kf)*32 + lane)*16 + h]
// ---------------------------------------------------------------------------
__global__ __launch_bounds__(256) void egcl_wfrag_kernel(
    const float* __restrict__ we2, const float* __restrict__ wc1,
    _Float16* __restrict__ frag2, _Float16* __restrict__ fragC)
{
    const int idx = blockIdx.x * blockDim.x + threadIdx.x;
    if (idx >= 2 * 16384) return;
    const float*  src = (idx < 16384) ? we2 : wc1;
    _Float16*     dst = (idx < 16384) ? frag2 : fragC;
    const int e    = idx & 16383;
    const int hh   = e & 15;
    const int lane = (e >> 4) & 31;
    const int kf   = (e >> 9) & 3;
    const int ct   = e >> 11;
    const int K    = kf * 32 + ((lane & 16) ? 8 : 0) + (hh & 7) + ((hh & 8) ? 16 : 0);
    const int n    = ct * 16 + (lane & 15);
    dst[e] = (_Float16)src[K * DD + n];
}

// ---------------------------------------------------------------------------
// Kernel 3: main edge kernel.  One workgroup per (b,i); 8 waves; each wave
// processes two 16-edge tiles (j blocks).  Produces agg_m[b,i,:] and x_out.
// ---------------------------------------------------------------------------
__global__ __launch_bounds__(256) void egcl_edge_kernel(
    const float* __restrict__ x, const int* __restrict__ mask,
    const float* __restrict__ Apre, const float* __restrict__ Bpre,
    const float* __restrict__ be1, const float* __restrict__ wlast,
    const _Float16* __restrict__ wfrag2, const _Float16* __restrict__ wfragC,
    const float* __restrict__ be2, const float* __restrict__ bc1,
    const float* __restrict__ wc2,
    float* __restrict__ aggout, float* __restrict__ xout)
{
    __shared__ __align__(16) _Float16 s_tile[8][16][PADF];
    __shared__ __align__(16) float s_Apre[DD];
    __shared__ __align__(16) float s_be1[DD];
    __shared__ __align__(16) float s_wl[DD];
    __shared__ float s_be2[DD];
    __shared__ float s_bc1[DD];
    __shared__ float s_wc2[DD];
    __shared__ float s_cm[8][16];
    __shared__ float s_d2[8][16];
    __shared__ float s_w[8][16];
    __shared__ float s_agg[DD];
    __shared__ float s_coord[3];

    const int bi   = blockIdx.x;          // b*N + i
    const int b    = bi >> 8;
    const int tid  = threadIdx.x;
    const int wave = tid >> 5;
    const int lane = tid & 31;
    const int colq = lane & 15;
    const int half8 = (lane >> 4) << 3;   // 0 or 8

    if (tid < DD) {
        s_Apre[tid] = Apre[bi * DD + tid];
        s_be1[tid]  = be1[tid];
        s_wl[tid]   = wlast[tid];
        s_be2[tid]  = be2[tid];
        s_bc1[tid]  = bc1[tid];
        s_wc2[tid]  = wc2[tid];
        s_agg[tid]  = 0.f;
    }
    if (tid < 3) s_coord[tid] = 0.f;

    const float xi0 = x[bi * 3 + 0];
    const float xi1 = x[bi * 3 + 1];
    const float xi2 = x[bi * 3 + 2];
    const float mi  = (float)mask[bi];

    float cacc0 = 0.f, cacc1 = 0.f, cacc2 = 0.f;
    __syncthreads();

#pragma unroll 1
    for (int tt = 0; tt < 2; ++tt) {
        const int j0 = (wave + tt * 8) * 16;

        // --- step 1: rel_x, dist^2, mask per edge (lanes 0..15 = edges) ---
        float rx0 = 0.f, rx1 = 0.f, rx2 = 0.f, cme = 0.f;
        if (lane < 16) {
            const int j = j0 + lane;
            rx0 = xi0 - x[(b * NN + j) * 3 + 0];
            rx1 = xi1 - x[(b * NN + j) * 3 + 1];
            rx2 = xi2 - x[(b * NN + j) * 3 + 2];
            const float d2 = rx0 * rx0 + rx1 * rx1 + rx2 * rx2 + 1e-8f;
            cme = mi * (float)mask[b * NN + j];
            s_d2[wave][lane] = d2;
            s_cm[wave][lane] = cme;
        }
        __syncthreads();

        // --- step 2: t1 = silu(Apre[i] + Bpre[j] + d2*wlast + be1) -> LDS f16
        {
            const int f0 = lane * 4;
            const float4 ap = *(const float4*)&s_Apre[f0];
            const float4 b1 = *(const float4*)&s_be1[f0];
            const float4 wl = *(const float4*)&s_wl[f0];
#pragma unroll 4
            for (int e = 0; e < 16; ++e) {
                const int j = j0 + e;
                const float4 bp = *(const float4*)&Bpre[(b * NN + j) * DD + f0];
                const float d2 = s_d2[wave][e];
                v4h hv;
                hv[0] = (_Float16)silu_f(ap.x + bp.x + d2 * wl.x + b1.x);
                hv[1] = (_Float16)silu_f(ap.y + bp.y + d2 * wl.y + b1.y);
                hv[2] = (_Float16)silu_f(ap.z + bp.z + d2 * wl.z + b1.z);
                hv[3] = (_Float16)silu_f(ap.w + bp.w + d2 * wl.w + b1.w);
                *(v4h*)&s_tile[wave][e][f0] = hv;
            }
        }
        __syncthreads();

        // --- step 3: hoist A fragments of t1 (CDNA5 16-bit A 16x32 layout) ---
        const int arow = lane & 15;
        const int aoff = (lane & 16) ? 8 : 0;
        v16h afrag[4];
#pragma unroll
        for (int kf = 0; kf < 4; ++kf) {
            const int koff = kf * 32 + aoff;
            afrag[kf] = combine16(*(const v8h*)&s_tile[wave][arow][koff],
                                  *(const v8h*)&s_tile[wave][arow][koff + 16]);
        }
        float cmr[8];
#pragma unroll
        for (int r = 0; r < 8; ++r) cmr[r] = s_cm[wave][r + half8];
        __syncthreads();   // t1 fully consumed; s_tile may be overwritten

        // --- step 4: GEMM1  m = silu(t1 @ we2 + be2) * cm  -> s_tile ---
        // unroll(1): keep exactly one ct's B-fragments + accumulator live.
#pragma unroll 1
        for (int ct = 0; ct < 8; ++ct) {
            v8f acc = {0.f, 0.f, 0.f, 0.f, 0.f, 0.f, 0.f, 0.f};
#pragma unroll
            for (int kf = 0; kf < 4; ++kf) {
                const _Float16* bp = wfrag2 + ((size_t)((ct * 4 + kf) * 32 + lane)) * 16;
                const v16h bfrag = combine16(*(const v8h*)bp, *(const v8h*)(bp + 8));
                acc = __builtin_amdgcn_wmma_f32_16x16x32_f16(
                    false, afrag[kf], false, bfrag, (short)0, acc, false, false);
            }
            const float be2c = s_be2[ct * 16 + colq];
            float rowsum = 0.f;
#pragma unroll
            for (int r = 0; r < 8; ++r) {
                const float v = silu_f(acc[r] + be2c) * cmr[r];
                rowsum += v;
                s_tile[wave][r + half8][ct * 16 + colq] = (_Float16)v;
            }
            atomicAdd(&s_agg[ct * 16 + colq], rowsum);
        }
        __syncthreads();

        // --- step 5: reload A fragments (now m), GEMM2 + wc2 dot ---
#pragma unroll
        for (int kf = 0; kf < 4; ++kf) {
            const int koff = kf * 32 + aoff;
            afrag[kf] = combine16(*(const v8h*)&s_tile[wave][arow][koff],
                                  *(const v8h*)&s_tile[wave][arow][koff + 16]);
        }
        float wacc[8];
#pragma unroll
        for (int r = 0; r < 8; ++r) wacc[r] = 0.f;
#pragma unroll 1
        for (int ct = 0; ct < 8; ++ct) {
            v8f acc = {0.f, 0.f, 0.f, 0.f, 0.f, 0.f, 0.f, 0.f};
#pragma unroll
            for (int kf = 0; kf < 4; ++kf) {
                const _Float16* bp = wfragC + ((size_t)((ct * 4 + kf) * 32 + lane)) * 16;
                const v16h bfrag = combine16(*(const v8h*)bp, *(const v8h*)(bp + 8));
                acc = __builtin_amdgcn_wmma_f32_16x16x32_f16(
                    false, afrag[kf], false, bfrag, (short)0, acc, false, false);
            }
            const float bc  = s_bc1[ct * 16 + colq];
            const float wcv = s_wc2[ct * 16 + colq];
#pragma unroll
            for (int r = 0; r < 8; ++r) {
                wacc[r] += silu_f(acc[r] + bc) * wcv;
            }
        }
        // reduce over the 16 lanes of each half (columns) -> per-edge dot
#pragma unroll
        for (int r = 0; r < 8; ++r) {
            float v = wacc[r];
            v += __shfl_xor(v, 1, 32);
            v += __shfl_xor(v, 2, 32);
            v += __shfl_xor(v, 4, 32);
            v += __shfl_xor(v, 8, 32);
            if ((lane & 15) == 0) s_w[wave][r + half8] = v;
        }
        __syncthreads();
        if (lane < 16) {
            const float w = tanhf(s_w[wave][lane]) * 0.1f * cme;
            cacc0 += rx0 * w;
            cacc1 += rx1 * w;
            cacc2 += rx2 * w;
        }
        __syncthreads();   // protect s_tile/s_d2/s_cm for next tile
    }

    // --- epilogue: coordinate reduction across lanes 0..15, then cross-wave ---
    float c0 = cacc0, c1 = cacc1, c2 = cacc2;
#pragma unroll
    for (int m = 1; m <= 8; m <<= 1) {
        c0 += __shfl_xor(c0, m, 32);
        c1 += __shfl_xor(c1, m, 32);
        c2 += __shfl_xor(c2, m, 32);
    }
    if (lane == 0) {
        atomicAdd(&s_coord[0], c0);
        atomicAdd(&s_coord[1], c1);
        atomicAdd(&s_coord[2], c2);
    }
    __syncthreads();
    if (tid < DD) aggout[bi * DD + tid] = s_agg[tid];
    if (tid < 3)  xout[bi * 3 + tid] = x[bi * 3 + tid] + s_coord[tid] * (1.0f / 256.0f);
}

// ---------------------------------------------------------------------------
// Kernel 4: node MLP + residual + LayerNorm.  One block (128 thr) per node.
// ---------------------------------------------------------------------------
__global__ __launch_bounds__(128) void egcl_node_kernel(
    const float* __restrict__ h, const float* __restrict__ agg,
    const float* __restrict__ wn1, const float* __restrict__ bn1,
    const float* __restrict__ wn2, const float* __restrict__ bn2,
    const float* __restrict__ gamma, const float* __restrict__ beta,
    float* __restrict__ hout)
{
    __shared__ float s_in[2 * DD];
    __shared__ float s_hid[DD];
    __shared__ float s_red[8];
    const int row = blockIdx.x;
    const int f   = threadIdx.x;
    const float hval = h[row * DD + f];
    s_in[f]      = hval;
    s_in[DD + f] = agg[row * DD + f];
    __syncthreads();
    float v = bn1[f];
#pragma unroll 8
    for (int k = 0; k < 2 * DD; ++k) v += s_in[k] * wn1[k * DD + f];
    s_hid[f] = silu_f(v);
    __syncthreads();
    float o = bn2[f];
#pragma unroll 8
    for (int k = 0; k < DD; ++k) o += s_hid[k] * wn2[k * DD + f];
    const float h2 = hval + o;
    // LayerNorm reduction (4 waves of 32)
    float s = h2, s2 = h2 * h2;
#pragma unroll
    for (int m = 1; m <= 16; m <<= 1) {
        s  += __shfl_xor(s,  m, 32);
        s2 += __shfl_xor(s2, m, 32);
    }
    const int wv = f >> 5, ln = f & 31;
    if (ln == 0) { s_red[wv] = s; s_red[4 + wv] = s2; }
    __syncthreads();
    const float tot  = s_red[0] + s_red[1] + s_red[2] + s_red[3];
    const float tot2 = s_red[4] + s_red[5] + s_red[6] + s_red[7];
    const float mu   = tot * (1.0f / 128.0f);
    const float var  = tot2 * (1.0f / 128.0f) - mu * mu;
    const float rstd = rsqrtf(var + 1e-5f);
    hout[row * DD + f] = (h2 - mu) * rstd * gamma[f] + beta[f];
}

// ---------------------------------------------------------------------------
extern "C" void kernel_launch(void* const* d_in, const int* in_sizes, int n_in,
                              void* d_out, int out_size, void* d_ws, size_t ws_size,
                              hipStream_t stream)
{
    (void)in_sizes; (void)n_in; (void)out_size; (void)ws_size;
    const float* h     = (const float*)d_in[0];
    const float* x     = (const float*)d_in[1];
    const int*   mask  = (const int*)d_in[2];
    const float* we1   = (const float*)d_in[3];
    const float* be1   = (const float*)d_in[4];
    const float* we2   = (const float*)d_in[5];
    const float* be2   = (const float*)d_in[6];
    const float* wn1   = (const float*)d_in[7];
    const float* bn1   = (const float*)d_in[8];
    const float* wn2   = (const float*)d_in[9];
    const float* bn2   = (const float*)d_in[10];
    const float* wc1   = (const float*)d_in[11];
    const float* bc1   = (const float*)d_in[12];
    const float* wc2   = (const float*)d_in[13];
    const float* gamma = (const float*)d_in[14];
    const float* beta  = (const float*)d_in[15];

    char* ws = (char*)d_ws;
    float*    Apre  = (float*)(ws);                                  // 1 MB
    float*    Bpre  = (float*)(ws + (size_t)(1 << 20));              // 1 MB
    float*    agg   = (float*)(ws + (size_t)(2 << 20));              // 1 MB
    _Float16* frag2 = (_Float16*)(ws + (size_t)(3 << 20));           // 32 KB
    _Float16* fragC = (_Float16*)(ws + (size_t)(3 << 20) + 32768);   // 32 KB

    float* hout = (float*)d_out;                 // [8,256,128]
    float* xout = hout + BATCH * NN * DD;        // [8,256,3]

    egcl_pre_kernel<<<BATCH * NN, 128, 0, stream>>>(h, we1, Apre, Bpre);
    egcl_wfrag_kernel<<<128, 256, 0, stream>>>(we2, wc1, frag2, fragC);
    egcl_edge_kernel<<<BATCH * NN, 256, 0, stream>>>(
        x, mask, Apre, Bpre, be1, we1 + 256 * DD, frag2, fragC,
        be2, bc1, wc2, agg, xout);
    egcl_node_kernel<<<BATCH * NN, 128, 0, stream>>>(
        h, agg, wn1, bn1, wn2, bn2, gamma, beta, hout);
}
